// WavefrontEngine_44744969290036
// MI455X (gfx1250) — matrix operand order repaired and checked by reading, not compiled, not run on hardware
//
#include <hip/hip_runtime.h>
#include <cstdint>
#include <cstddef>

// ---------------------------------------------------------------------------
// Problem constants (from reference): L=6 layers, S=4096, D=3 deps, P=2 ports,
// B=8 batch, DIM=128.  out = 2 grids of (L*S, B, DIM) f32.
// ---------------------------------------------------------------------------
#define L_   6
#define S_   4096
#define D_   3
#define P_   2
#define B_   8
#define DIM_ 128

#define ROWS_   (S_ * B_)                 // 32768 rows per layer (M)
#define GELEMS_ (S_ * B_ * DIM_)          // elems in one (S,B,DIM) activation
#define OUTP_STRIDE_ ((size_t)L_ * S_ * B_ * DIM_)  // per-port out stride

typedef __attribute__((ext_vector_type(16))) __bf16 v16bf;
typedef __attribute__((ext_vector_type(8)))  float  v8f;

struct Frag32 { uint4 lo, hi; };          // 32 bytes == v16bf

// ---------------------------------------------------------------------------
// Prep kernel 1: x (B,S,DIM) f32  ->  xb (S,B,DIM) bf16
// ---------------------------------------------------------------------------
__global__ __launch_bounds__(256) void wf_prep_x(const float* __restrict__ x,
                                                 __bf16* __restrict__ xb) {
  int tid = blockIdx.x * blockDim.x + threadIdx.x;   // S_*B_*DIM_ threads
  int o = tid & (DIM_ - 1);
  int b = (tid >> 7) & (B_ - 1);
  int s = tid >> 10;
  xb[tid] = (__bf16)x[((size_t)b * S_ + s) * DIM_ + o];
}

// ---------------------------------------------------------------------------
// Prep kernel 2: swizzle W (L,D,P,K=128,N=128) f32 into per-lane B-fragment
// bf16 layout.  Fragment (d,i,p,kc,nb) = 512 halves:
//   half index = lane*16 + j ;  K = kc*32 + (lane>=16 ? 16 : 0) + j ,
//   N = nb*16 + (lane&15)        (contiguous-K order used here, same as A frags)
// ---------------------------------------------------------------------------
__global__ __launch_bounds__(256) void wf_prep_w(const float* __restrict__ W,
                                                 __bf16* __restrict__ wsw) {
  int tid = blockIdx.x * blockDim.x + threadIdx.x;   // 1152*512 threads
  int j    = tid & 15;
  int lane = (tid >> 4) & 31;
  int frag = tid >> 9;
  int nb = frag & 7;
  int kc = (frag >> 3) & 3;
  int p  = (frag >> 5) & 1;
  int rest = frag >> 6;
  int i = rest % 3;
  int d = rest / 3;
  int K = kc * 32 + ((lane & 16) ? 16 : 0) + j;
  int O = nb * 16 + (lane & 15);
  wsw[tid] = (__bf16)W[(((size_t)(d * 3 + i) * 2 + p) * DIM_ + K) * DIM_ + O];
}

// ---------------------------------------------------------------------------
// Layer kernel: one wave computes a (16-row M-tile) x (all 128 N) for ONE port.
// 4 K-chunks x 8 N-tiles x 3 deps = 96 v_wmma_f32_16x16x32_bf16 per wave.
// ---------------------------------------------------------------------------
__global__ __launch_bounds__(256) void wf_layer(
    int d,
    const float*  __restrict__ bias,   // (L,P,DIM) f32
    const __bf16* __restrict__ wsw,    // swizzled W fragments
    const __bf16* __restrict__ a0,     // dep0 source (S,B,DIM) bf16
    const __bf16* __restrict__ a1,     // dep1 source
    const __bf16* __restrict__ a2,     // dep2 source
    __bf16* __restrict__ g0w,          // this layer's bf16 activations, port0
    __bf16* __restrict__ g1w,          // port1
    float*  __restrict__ out)          // full f32 output base
{
  const int lane = threadIdx.x & 31;
  const int wave = blockIdx.x * (blockDim.x >> 5) + (threadIdx.x >> 5);
  const int p    = wave & 1;           // port
  const int tile = wave >> 1;          // 0..2047  (16-row M tile)
  const int s0   = tile * 2;           // 16 rows == 2 s-values x 8 batches

  // This lane's A-row (M = lane&15): s, b and the ks0 sub-chunk offset.
  const int m   = lane & 15;
  const int su  = s0 + (m >> 3);
  const int bb  = m & 7;
  const int ks0 = (lane & 16) ? 16 : 0;

  // Per-dep A row pointers + boundary validity (value is UNshifted at d==0,
  // but boundary masking always follows the shift -- matches reference rm).
  const __bf16* arow[3];
  bool avalid[3];
  const int shifts[3] = {0, -1, 1};
  const __bf16* asrc[3] = {a0, a1, a2};
#pragma unroll
  for (int i = 0; i < 3; ++i) {
    int sv = su + shifts[i];
    avalid[i] = (sv >= 0) && (sv < S_);
    int srow = (d == 0) ? su : sv;
    arow[i] = asrc[i] + ((size_t)srow * B_ + bb) * DIM_;
  }

  const __bf16* wlane = wsw + (size_t)lane * 16;

  v8f acc[8];
#pragma unroll
  for (int nb = 0; nb < 8; ++nb) {
    v8f z = {0.f, 0.f, 0.f, 0.f, 0.f, 0.f, 0.f, 0.f};
    acc[nb] = z;
  }

#pragma unroll
  for (int kc = 0; kc < 4; ++kc) {
    // A fragments: contiguous K per lane (matches swizzled B):
    //   halves 0..15 : K = kc*32 + ks0 + j
    v16bf afrag[3];
#pragma unroll
    for (int i = 0; i < 3; ++i) {
      Frag32 f;
      f.lo = make_uint4(0u, 0u, 0u, 0u);
      f.hi = make_uint4(0u, 0u, 0u, 0u);
      if (avalid[i]) {
        const __bf16* pr = arow[i] + kc * 32 + ks0;
        f.lo = *(const uint4*)(pr);        // 8 halves
        f.hi = *(const uint4*)(pr + 8);    // 8 halves, +16 bytes
      }
      afrag[i] = __builtin_bit_cast(v16bf, f);
    }
#pragma unroll
    for (int nb = 0; nb < 8; ++nb) {
#pragma unroll
      for (int i = 0; i < 3; ++i) {
        int fragidx = ((d * 3 + i) * 2 + p) * 32 + kc * 8 + nb;
        v16bf bfrag = *(const v16bf*)(wlane + (size_t)fragidx * 512);
        acc[nb] = __builtin_amdgcn_wmma_f32_16x16x32_bf16(
            /*neg_a=*/false, afrag[i],
            /*neg_b=*/false, bfrag,
            /*c_mod=*/(short)0, acc[nb],
            /*reuse_a=*/false, /*reuse_b=*/false);
      }
    }
  }

  // Epilogue: bias + tanh; f32 to d_out, bf16 to ping-pong grid.
  // C/D layout: VGPR r, lanes 0-15 -> M=r ; lanes 16-31 -> M=r+8 ; N=lane&15.
  const int onl = lane & 15;
  __bf16* gw = p ? g1w : g0w;
  float* outp = out + (size_t)p * OUTP_STRIDE_ + (size_t)d * GELEMS_;
#pragma unroll
  for (int nb = 0; nb < 8; ++nb) {
    int o = nb * 16 + onl;
    float bv = bias[((size_t)d * P_ + p) * DIM_ + o];
#pragma unroll
    for (int vr = 0; vr < 8; ++vr) {
      int mr = vr + ((lane & 16) ? 8 : 0);
      int s  = s0 + (mr >> 3);
      int b  = mr & 7;
      float val = tanhf(acc[nb][vr] + bv);
      size_t ridx = ((size_t)s * B_ + b) * DIM_ + o;
      outp[ridx] = val;
      gw[ridx] = (__bf16)val;
    }
  }
}

// ---------------------------------------------------------------------------
// Workspace layout (bytes):
//   [0, 1.18MB)   swizzled W (bf16)
//   [2MB, 10MB)   xb (S,B,DIM) bf16
//   [10MB, 42MB)  4 activation buffers: slot{0,1} x port{0,1}, 8MB each
// ---------------------------------------------------------------------------
#define WS_OFF_W  ((size_t)0)
#define WS_OFF_XB ((size_t)(2u << 20))
#define WS_OFF_G  ((size_t)(10u << 20))
#define WS_GBUF   ((size_t)GELEMS_ * 2)   // 8,388,608 bytes

extern "C" void kernel_launch(void* const* d_in, const int* in_sizes, int n_in,
                              void* d_out, int out_size, void* d_ws, size_t ws_size,
                              hipStream_t stream) {
  (void)in_sizes; (void)n_in; (void)out_size; (void)ws_size;

  const float* x  = (const float*)d_in[0];   // (B,S,DIM) f32
  const float* W  = (const float*)d_in[1];   // (L,D,P,DIM,DIM) f32
  const float* bi = (const float*)d_in[2];   // (L,P,DIM) f32
  // d_in[3]: routing_map -- routing is statically known, unused.
  float* out = (float*)d_out;

  char* ws = (char*)d_ws;
  __bf16* wsw = (__bf16*)(ws + WS_OFF_W);
  __bf16* xb  = (__bf16*)(ws + WS_OFF_XB);
  __bf16* g[2][2];
  for (int slot = 0; slot < 2; ++slot)
    for (int port = 0; port < 2; ++port)
      g[slot][port] = (__bf16*)(ws + WS_OFF_G + ((size_t)(slot * 2 + port)) * WS_GBUF);

  // Prep: x transpose+convert, W swizzle+convert.
  wf_prep_x<<<(S_ * B_ * DIM_) / 256, 256, 0, stream>>>(x, xb);
  wf_prep_w<<<(L_ * D_ * P_ * DIM_ * DIM_) / 256, 256, 0, stream>>>(W, wsw);

  // Sequential layers (layer d depends on layer d-1).
  for (int d = 0; d < L_; ++d) {
    int wslot = d & 1;
    int rslot = wslot ^ 1;
    const __bf16 *a0, *a1, *a2;
    if (d == 0) { a0 = xb; a1 = xb; a2 = xb; }
    else        { a0 = g[rslot][0]; a1 = g[rslot][0]; a2 = g[rslot][1]; }
    // 2048 M-tiles x 2 ports = 4096 waves ; 8 waves/block -> 512 blocks.
    wf_layer<<<512, 256, 0, stream>>>(d, bi, wsw, a0, a1, a2,
                                      g[wslot][0], g[wslot][1], out);
  }
}
